// InstanceLoss_37314675867760
// MI455X (gfx1250) — matrix operand order repaired and checked by reading, not compiled, not run on hardware
//
#include <hip/hip_runtime.h>
#include <hip/hip_bf16.h>
#include <math.h>

#define P_TOT (512 * 512)
#define C_CH  64
#define K_INST 20
#define NCLS  11
#define KPAD  32
#define ROWS_A 80      // 64 channel rows + [64]=sum_p m*sum_c v1^2 (=sq), [65]=counts, rest junk
#define ROWS_B 64
#define ELEMS_A (ROWS_A * KPAD)   // 2560
#define ELEMS_B (ROWS_B * KPAD)   // 2048
#define NBLK 64
#define NTHR 256

typedef __attribute__((ext_vector_type(16))) __bf16 v16bf;
typedef __attribute__((ext_vector_type(8)))  float  v8f;

__global__ void zero_ws_kernel(float* __restrict__ ws, int n) {
    int i = blockIdx.x * blockDim.x + threadIdx.x;
    if (i < n) ws[i] = 0.0f;
}

// Deterministic cross-block reduction of per-block partial tiles.
__global__ void reduce_partials_kernel(const float* __restrict__ part,
                                       float* __restrict__ outp,
                                       int elems, int nslices) {
    int i = blockIdx.x * blockDim.x + threadIdx.x;
    if (i < elems) {
        float a = 0.0f;
        for (int b = 0; b < nslices; ++b) a += part[(size_t)b * elems + i];
        outp[i] = a;
    }
}

// WHICH==0: A = v1 channels + extra tile (row0 = per-pixel sum_c v1^2, row1 = 1.0)
// WHICH==1: A = v2 / max(||v2[:,p]||, 1e-8)
// B = mask^T (bf16 0/1, exact). out'[row, k] = sum_p A[row,p]*m[k,p], f32 accum.
// ATOMIC==false: block writes its private partial slice (plain stores, deterministic).
// ATOMIC==true : block atomically adds its partial into the final array (fallback).
template<int WHICH, bool ATOMIC>
__global__ __launch_bounds__(NTHR)
void masked_gemm_kernel(const float* __restrict__ V,
                        const int*   __restrict__ Msk,
                        float* __restrict__ outBase)
{
    constexpr int NTM = (WHICH == 0) ? 5 : 4;
    constexpr int NEL = NTM * 16 * KPAD;

    __shared__ float s_acc[NEL];

    const int tid   = threadIdx.x;
    const int lane  = tid & 31;
    const int half  = lane >> 4;      // which 16-lane half
    const int lr    = lane & 15;      // lane within half
    // wave-uniform scalars -> scalar loop control (EXEC stays all-1s for WMMA)
    const int waveInBlk = __builtin_amdgcn_readfirstlane(tid >> 5);
    const int gwave  = blockIdx.x * (NTHR >> 5) + waveInBlk;
    const int nwaves = NBLK * (NTHR >> 5);

    for (int i = tid; i < NEL; i += NTHR) s_acc[i] = 0.0f;
    __syncthreads();

    v8f acc[NTM][2];
    #pragma unroll
    for (int tm = 0; tm < NTM; ++tm)
        #pragma unroll
        for (int tn = 0; tn < 2; ++tn)
            #pragma unroll
            for (int r = 0; r < 8; ++r)
                acc[tm][tn][r] = 0.0f;

    const int nchunk  = P_TOT / 32;
    const int pstride = nwaves * 32;

    for (int chunk = gwave; chunk < nchunk; chunk += nwaves) {
        const int p0 = chunk * 32;
        // ISA 16-bit A 16x32 layout: elems 0..7 -> pA+0..7, elems 8..15 -> pA+16..23
        const int pA = p0 + half * 8;

        float af[4][16];
        #pragma unroll
        for (int t = 0; t < 4; ++t) {
            const float* src = V + (size_t)(t * 16 + lr) * P_TOT + pA;
            __builtin_prefetch(src + pstride, 0, 0);   // global_prefetch_b8 stream hint
            float4 f0 = *(const float4*)(src + 0);
            float4 f1 = *(const float4*)(src + 4);
            float4 f2 = *(const float4*)(src + 16);
            float4 f3 = *(const float4*)(src + 20);
            af[t][0]=f0.x;  af[t][1]=f0.y;  af[t][2]=f0.z;  af[t][3]=f0.w;
            af[t][4]=f1.x;  af[t][5]=f1.y;  af[t][6]=f1.z;  af[t][7]=f1.w;
            af[t][8]=f2.x;  af[t][9]=f2.y;  af[t][10]=f2.z; af[t][11]=f2.w;
            af[t][12]=f3.x; af[t][13]=f3.y; af[t][14]=f3.z; af[t][15]=f3.w;
        }

        // per-pixel sum over channels of v^2: each half-wave holds the same 16
        // pixels across 16 channel-rows per tile -> butterfly over lanes 0..15.
        float s[16];
        #pragma unroll
        for (int i = 0; i < 16; ++i) {
            float q = 0.0f;
            #pragma unroll
            for (int t = 0; t < 4; ++t) q += af[t][i] * af[t][i];
            s[i] = q;
        }
        #pragma unroll
        for (int off = 1; off < 16; off <<= 1)
            #pragma unroll
            for (int i = 0; i < 16; ++i)
                s[i] += __shfl_xor(s[i], off, 32);

        if (WHICH == 1) {
            #pragma unroll
            for (int i = 0; i < 16; ++i) {
                const float inv = 1.0f / fmaxf(sqrtf(s[i]), 1e-8f);
                #pragma unroll
                for (int t = 0; t < 4; ++t) af[t][i] *= inv;
            }
        }

        v16bf a[NTM];
        #pragma unroll
        for (int t = 0; t < 4; ++t)
            #pragma unroll
            for (int i = 0; i < 16; ++i)
                a[t][i] = (__bf16)af[t][i];
        if (WHICH == 0) {
            #pragma unroll
            for (int i = 0; i < 16; ++i) {
                const float rv = (lr == 0) ? s[i] : ((lr == 1) ? 1.0f : 0.0f);
                a[4][i] = (__bf16)rv;
            }
        }

        // B fragment (bf16 B 32x16): same K order as A: elems 0..7 -> pB+0..7, elems 8..15 -> pB+16..23
        v16bf b[2];
        const int pB = p0 + half * 8;
        #pragma unroll
        for (int tn = 0; tn < 2; ++tn) {
            const int k = tn * 16 + lr;
            float mv[16];
            if (k < K_INST) {
                const int* ms = Msk + (size_t)k * P_TOT + pB;
                int4 m0 = *(const int4*)(ms + 0);
                int4 m1 = *(const int4*)(ms + 4);
                int4 m2 = *(const int4*)(ms + 16);
                int4 m3 = *(const int4*)(ms + 20);
                mv[0]=(float)m0.x;  mv[1]=(float)m0.y;  mv[2]=(float)m0.z;  mv[3]=(float)m0.w;
                mv[4]=(float)m1.x;  mv[5]=(float)m1.y;  mv[6]=(float)m1.z;  mv[7]=(float)m1.w;
                mv[8]=(float)m2.x;  mv[9]=(float)m2.y;  mv[10]=(float)m2.z; mv[11]=(float)m2.w;
                mv[12]=(float)m3.x; mv[13]=(float)m3.y; mv[14]=(float)m3.z; mv[15]=(float)m3.w;
            } else {
                #pragma unroll
                for (int i = 0; i < 16; ++i) mv[i] = 0.0f;
            }
            #pragma unroll
            for (int i = 0; i < 16; ++i) b[tn][i] = (__bf16)mv[i];
        }

        #pragma unroll
        for (int tm = 0; tm < NTM; ++tm)
            #pragma unroll
            for (int tn = 0; tn < 2; ++tn)
                acc[tm][tn] = __builtin_amdgcn_wmma_f32_16x16x32_bf16(
                    false, a[tm], false, b[tn], (short)0, acc[tm][tn], false, false);
    }

    // Intra-block combine: D layout VGPR r, lanes 0-15 -> M=r, lanes 16-31 -> M=r+8; N=lane%16
    #pragma unroll
    for (int tm = 0; tm < NTM; ++tm)
        #pragma unroll
        for (int tn = 0; tn < 2; ++tn)
            #pragma unroll
            for (int r = 0; r < 8; ++r) {
                const int row = tm * 16 + r + half * 8;
                const int col = tn * 16 + lr;
                atomicAdd(&s_acc[row * KPAD + col], acc[tm][tn][r]);  // ds_add_f32
            }
    __syncthreads();

    if (ATOMIC) {
        for (int i = tid; i < NEL; i += NTHR) atomicAdd(&outBase[i], s_acc[i]);
    } else {
        float* dst = outBase + (size_t)blockIdx.x * NEL;
        for (int i = tid; i < NEL; i += NTHR) dst[i] = s_acc[i];
    }
}

__global__ __launch_bounds__(256)
void finalize_kernel(const float* __restrict__ wsA, const float* __restrict__ wsB,
                     const int* __restrict__ labels, float* __restrict__ out)
{
    __shared__ float s_m1[K_INST][C_CH];   // mean1 then mean1n
    __shared__ float s_u[K_INST][C_CH];
    __shared__ float s_cnt[K_INST];
    __shared__ float s_std[K_INST];
    __shared__ float s_sim[K_INST][K_INST];
    const int tid = threadIdx.x;

    if (tid < K_INST) s_cnt[tid] = wsA[65 * KPAD + tid];   // counts row
    __syncthreads();

    for (int idx = tid; idx < K_INST * C_CH; idx += blockDim.x) {
        const int k = idx / C_CH, c = idx % C_CH;
        const float cnt = s_cnt[k];
        s_m1[k][c] = wsA[c * KPAD + k] / cnt;   // mean1
        s_u[k][c]  = wsB[c * KPAD + k] / cnt;   // masked mean of unit v2 pixels
    }
    __syncthreads();

    if (tid < K_INST) {
        const float cnt = s_cnt[tid];
        float ssum = 0.0f, nrm = 0.0f;
        for (int c = 0; c < C_CH; ++c) { const float v = s_m1[tid][c]; ssum += v; nrm += v * v; }
        const float n_elem = cnt * (float)C_CH;
        const float em = ssum / (float)C_CH;            // sum1_total/(cnt*C) since m1 = sum1/cnt
        const float sq = wsA[64 * KPAD + tid];          // sum_p m * sum_c v1^2
        const float var = (sq - n_elem * em * em) / (n_elem - 1.0f);
        s_std[tid] = sqrtf(fmaxf(var, 0.0f));
        const float inv = 1.0f / fmaxf(sqrtf(nrm), 1e-8f);
        for (int c = 0; c < C_CH; ++c) s_m1[tid][c] *= inv;   // mean1n
    }
    __syncthreads();

    for (int idx = tid; idx < K_INST * K_INST; idx += blockDim.x) {
        const int k = idx / K_INST, j = idx % K_INST;
        float d = 0.0f;
        for (int c = 0; c < C_CH; ++c) d += s_m1[k][c] * s_u[j][c];
        s_sim[k][j] = d;
    }
    __syncthreads();

    if (tid == 0) {
        float inst[NCLS], csim[NCLS], cstd[NCLS], csum[NCLS];
        for (int c = 0; c < NCLS; ++c) { inst[c]=0.f; csim[c]=0.f; cstd[c]=0.f; csum[c]=0.f; }
        int cls[K_INST];
        for (int k = 0; k < K_INST; ++k) cls[k] = labels[k];
        float negsum = 0.0f, negcnt = 0.0f;
        for (int k = 0; k < K_INST; ++k) {
            inst[cls[k]] += s_sim[k][k];
            cstd[cls[k]] += s_std[k];
            csum[cls[k]] += 1.0f;
        }
        for (int k = 0; k < K_INST; ++k)
            for (int j = 0; j < K_INST; ++j) {
                if (j == k) continue;
                if (cls[j] == cls[k]) csim[cls[k]] += s_sim[k][j];
                else { negsum += s_sim[k][j]; negcnt += 1.0f; }
            }
        for (int c = 0; c < NCLS; ++c) {
            const bool g = csum[c] > 1.0f;
            const float safe = g ? csum[c] : 1.0f;
            const float pair = g ? csum[c] * (csum[c] - 1.0f) : 1.0f;
            out[c]                = inst[c] / safe;   // instance_sim
            out[NCLS + c]         = csim[c] / pair;   // class_sim
            out[2 * NCLS + 1 + c] = cstd[c] / safe;   // class_std
        }
        out[2 * NCLS] = negsum / negcnt;              // neg_sim
    }
}

extern "C" void kernel_launch(void* const* d_in, const int* in_sizes, int n_in,
                              void* d_out, int out_size, void* d_ws, size_t ws_size,
                              hipStream_t stream)
{
    const float* v1     = (const float*)d_in[0];
    const float* v2     = (const float*)d_in[1];
    const int*   labels = (const int*)d_in[2];
    const int*   masks  = (const int*)d_in[3];
    float* out = (float*)d_out;
    float* ws  = (float*)d_ws;

    const size_t needTwoPhase =
        ((size_t)NBLK * (ELEMS_A + ELEMS_B) + ELEMS_A + ELEMS_B) * sizeof(float);

    if (ws_size >= needTwoPhase) {
        // Deterministic two-phase: private partial slices + serial per-element reduce, no global atomics.
        float* partA = ws;
        float* partB = partA + (size_t)NBLK * ELEMS_A;
        float* finA  = partB + (size_t)NBLK * ELEMS_B;
        float* finB  = finA + ELEMS_A;
        masked_gemm_kernel<0, false><<<NBLK, NTHR, 0, stream>>>(v1, masks, partA);
        masked_gemm_kernel<1, false><<<NBLK, NTHR, 0, stream>>>(v2, masks, partB);
        reduce_partials_kernel<<<(ELEMS_A + 255) / 256, 256, 0, stream>>>(partA, finA, ELEMS_A, NBLK);
        reduce_partials_kernel<<<(ELEMS_B + 255) / 256, 256, 0, stream>>>(partB, finB, ELEMS_B, NBLK);
        finalize_kernel<<<1, 256, 0, stream>>>(finA, finB, labels, out);
    } else {
        // Fallback: per-block LDS pre-reduction, then 64 blocks of global atomics.
        float* finA = ws;
        float* finB = finA + ELEMS_A;
        const int nAcc = ELEMS_A + ELEMS_B;
        zero_ws_kernel<<<(nAcc + 255) / 256, 256, 0, stream>>>(ws, nAcc);
        masked_gemm_kernel<0, true><<<NBLK, NTHR, 0, stream>>>(v1, masks, finA);
        masked_gemm_kernel<1, true><<<NBLK, NTHR, 0, stream>>>(v2, masks, finB);
        finalize_kernel<<<1, 256, 0, stream>>>(finA, finB, labels, out);
    }
}
